// ChainClassifier_6554120094161
// MI455X (gfx1250) — matrix operand order repaired and checked by reading, not compiled, not run on hardware
//
#include <hip/hip_runtime.h>

typedef __bf16 bf16_t;
typedef __attribute__((ext_vector_type(16))) __bf16 v16bf;
typedef __attribute__((ext_vector_type(8)))  __bf16 v8bf;
typedef __attribute__((ext_vector_type(8)))  float  v8f;

#define NNOTES 88
#define XDIM   512
#define EMB    256
#define CL     12
#define DTOT   524            // XDIM + CL
#define BSTOT  8192           // B*S
#define MT     64             // rows (positions) per workgroup
#define XSTRIDE 520           // padded bf16 LDS stride for x tile
#define LSTRIDE 40            // padded bf16 LDS stride for l buffer
#define KB_CNT  17            // 16 x-blocks + 1 l-block (K padded 524 -> 544)
#define FRAG_ELEMS 512        // one B fragment: 32 lanes x 16 bf16
#define NOTE_FRAGS (KB_CNT * 16)            // 272 fragments per note
#define W1BF_ELEMS ((size_t)NNOTES * NOTE_FRAGS * FRAG_ELEMS)   // 12,255,232 bf16

static __device__ inline v8f v8f_zero() {
  v8f z;
#pragma unroll
  for (int i = 0; i < 8; ++i) z[i] = 0.0f;
  return z;
}

// ---------------------------------------------------------------------------
// Phase A: convert W1 (f32, [88][256][524]) into fragment-major bf16 tiles
// layout [n][kb][nb][lane][16], zero-padded in K to 544.
// ---------------------------------------------------------------------------
__global__ __launch_bounds__(256)
void convert_w1_kernel(const float* __restrict__ W1, bf16_t* __restrict__ w1bf) {
  size_t i = (size_t)blockIdx.x * 256 + threadIdx.x;
  if (i >= W1BF_ELEMS) return;
  int ii   = (int)(i & 15);
  int lane = (int)((i >> 4) & 31);
  size_t f = i >> 9;                 // fragment index = (n*17 + kb)*16 + nb
  int nb = (int)(f & 15);
  size_t g = f >> 4;                 // n*17 + kb
  int kb = (int)(g % KB_CNT);
  int n  = (int)(g / KB_CNT);
  int e  = nb * 16 + (lane & 15);
  int d  = kb * 32 + (ii < 8 ? (lane >> 4) * 8 + ii : 16 + (lane >> 4) * 8 + (ii - 8));
  float v = (d < DTOT) ? W1[((size_t)n * EMB + e) * DTOT + d] : 0.0f;
  w1bf[i] = (bf16_t)v;
}

// ---------------------------------------------------------------------------
// Main kernel (pre-converted bf16 W1): per-position 88-note chain,
// bf16 WMMA for a = h @ W1n^T, epilogue GEMV folded in registers.
// ---------------------------------------------------------------------------
__global__ __launch_bounds__(128)
void chain_wmma_pre(const float* __restrict__ x,
                    const float* __restrict__ Wc1, const float* __restrict__ bc1,
                    const float* __restrict__ Wc2, const float* __restrict__ bc2,
                    const bf16_t* __restrict__ w1bf, const float* __restrict__ b1,
                    const float* __restrict__ W2,  const float* __restrict__ b2,
                    const float* __restrict__ noise,
                    float* __restrict__ labels, float* __restrict__ pmass)
{
  __shared__ bf16_t xl[MT * XSTRIDE];
  __shared__ bf16_t lb[MT * LSTRIDE];
  __shared__ float  win[MT][CL];
  __shared__ float  logits[MT];
  __shared__ float  wc1s[CL * CL], wc2s[CL * CL], bc1s[CL], bc2s[CL];

  const int tid  = threadIdx.x;
  const int wave = tid >> 5;
  const int lane = tid & 31;
  const int m16  = lane & 15;
  const int hi   = lane >> 4;
  const int rowbase = blockIdx.x * MT;

  for (int i = tid; i < MT * XDIM; i += 128) {
    int r = i >> 9, c = i & 511;
    xl[r * XSTRIDE + c] = (bf16_t)x[(size_t)(rowbase + r) * XDIM + c];
  }
  for (int i = tid; i < MT * LSTRIDE; i += 128) lb[i] = (bf16_t)0.0f;
  for (int i = tid; i < MT * CL; i += 128) win[i / CL][i % CL] = 0.0f;
  for (int i = tid; i < CL * CL; i += 128) { wc1s[i] = Wc1[i]; wc2s[i] = Wc2[i]; }
  if (tid < CL) { bc1s[tid] = bc1[tid]; bc2s[tid] = bc2[tid]; }

  float pm = 0.0f;
  __syncthreads();

  for (int n = 0; n < NNOTES; ++n) {
    // ---- chain MLP: l = Wc2 @ relu(Wc1 @ window + bc1) + bc2 ----
    if (tid < MT) {
      float w[CL], l1[CL];
#pragma unroll
      for (int k = 0; k < CL; ++k) w[k] = win[tid][k];
#pragma unroll
      for (int j = 0; j < CL; ++j) {
        float s = bc1s[j];
#pragma unroll
        for (int k = 0; k < CL; ++k) s += wc1s[j * CL + k] * w[k];
        l1[j] = fmaxf(s, 0.0f);
      }
#pragma unroll
      for (int j = 0; j < CL; ++j) {
        float s = bc2s[j];
#pragma unroll
        for (int k = 0; k < CL; ++k) s += wc2s[j * CL + k] * l1[k];
        lb[tid * LSTRIDE + j] = (bf16_t)s;
      }
    }
    __syncthreads();

    // ---- WMMA GEMM with register double-buffered B fragments ----
    v8f acc[16];
#pragma unroll
    for (int t = 0; t < 16; ++t) acc[t] = v8f_zero();

    const bf16_t* fb = w1bf + (size_t)n * (NOTE_FRAGS * FRAG_ELEMS) + lane * 16;
    v16bf bcur = *(const v16bf*)fb;

    for (int kb = 0; kb < KB_CNT; ++kb) {
      // A fragment: two 16B LDS chunks per the ISA 16-bit A 16x32 layout
      v16bf a;
      if (kb < 16) {
        const bf16_t* ap = &xl[(wave * 16 + m16) * XSTRIDE + kb * 32 + hi * 8];
        v8bf c0 = *(const v8bf*)ap;
        v8bf c1 = *(const v8bf*)(ap + 16);
#pragma unroll
        for (int i = 0; i < 8; ++i) { a[i] = c0[i]; a[i + 8] = c1[i]; }
      } else {
        const bf16_t* ap = &lb[(wave * 16 + m16) * LSTRIDE + hi * 8];
        v8bf c0 = *(const v8bf*)ap;
        v8bf c1 = *(const v8bf*)(ap + 16);
#pragma unroll
        for (int i = 0; i < 8; ++i) { a[i] = c0[i]; a[i + 8] = c1[i]; }
      }

      const bf16_t* rowp = fb + (size_t)kb * (16 * FRAG_ELEMS);
#pragma unroll
      for (int nb = 0; nb < 16; ++nb) {
        const bf16_t* np;
        if (nb < 15)          np = rowp + (nb + 1) * FRAG_ELEMS;   // imm offsets
        else if (kb < 16)     np = rowp + 16 * FRAG_ELEMS;         // next kb row
        else                  np = fb;                             // dummy (last)
        v16bf bn = *(const v16bf*)np;
        acc[nb] = __builtin_amdgcn_wmma_f32_16x16x32_bf16(
            false, a, false, bcur, (short)0, acc[nb], false, false);
        bcur = bn;
      }
    }

    if (n + 1 < NNOTES)   // keep next note's tiles warm (gfx1250 prefetch path)
      __builtin_prefetch(w1bf + (size_t)(n + 1) * (NOTE_FRAGS * FRAG_ELEMS)
                              + (size_t)tid * 1024, 0, 1);

    // ---- epilogue: logit = relu(acc + b1) . w2, cross-lane reduce ----
    float part[8];
#pragma unroll
    for (int r = 0; r < 8; ++r) part[r] = 0.0f;
#pragma unroll
    for (int nb = 0; nb < 16; ++nb) {
      int e = nb * 16 + m16;
      float b1v = b1[n * EMB + e];
      float w2v = W2[n * EMB + e];
#pragma unroll
      for (int r = 0; r < 8; ++r) {
        float av = fmaxf(acc[nb][r] + b1v, 0.0f);
        part[r] += av * w2v;
      }
    }
#pragma unroll
    for (int r = 0; r < 8; ++r) {
#pragma unroll
      for (int off = 1; off < 16; off <<= 1)
        part[r] += __shfl_xor(part[r], off, 32);
    }
    if (m16 == 0) {
      float b2v = b2[n];
#pragma unroll
      for (int r = 0; r < 8; ++r)
        logits[wave * 16 + hi * 8 + r] = part[r] + b2v;
    }
    __syncthreads();

    // ---- sample, prob-mass, window shift, emit label ----
    if (tid < MT) {
      int gp = rowbase + tid;
      float logit = logits[tid];
      float prob  = 1.0f / (1.0f + __expf(-logit));
      float u     = noise[(size_t)n * BSTOT + gp];
      float pred  = (u < prob) ? 1.0f : 0.0f;
      pm += prob * pred + (1.0f - prob) * (1.0f - pred);
#pragma unroll
      for (int k = 0; k < CL - 1; ++k) win[tid][k] = win[tid][k + 1];
      win[tid][CL - 1] = pred;
      labels[(size_t)gp * NNOTES + n] = pred;
    }
  }

  if (tid < MT) pmass[rowbase + tid] = pm;
}

// ---------------------------------------------------------------------------
// Fallback (no workspace): inline f32->bf16 conversion of B fragments.
// ---------------------------------------------------------------------------
__global__ __launch_bounds__(128)
void chain_wmma_fb(const float* __restrict__ x,
                   const float* __restrict__ Wc1, const float* __restrict__ bc1,
                   const float* __restrict__ Wc2, const float* __restrict__ bc2,
                   const float* __restrict__ W1,  const float* __restrict__ b1,
                   const float* __restrict__ W2,  const float* __restrict__ b2,
                   const float* __restrict__ noise,
                   float* __restrict__ labels, float* __restrict__ pmass)
{
  __shared__ bf16_t xl[MT * XSTRIDE];
  __shared__ bf16_t lb[MT * LSTRIDE];
  __shared__ float  win[MT][CL];
  __shared__ float  logits[MT];
  __shared__ float  wc1s[CL * CL], wc2s[CL * CL], bc1s[CL], bc2s[CL];

  const int tid  = threadIdx.x;
  const int wave = tid >> 5;
  const int lane = tid & 31;
  const int m16  = lane & 15;
  const int hi   = lane >> 4;
  const int rowbase = blockIdx.x * MT;

  for (int i = tid; i < MT * XDIM; i += 128) {
    int r = i >> 9, c = i & 511;
    xl[r * XSTRIDE + c] = (bf16_t)x[(size_t)(rowbase + r) * XDIM + c];
  }
  for (int i = tid; i < MT * LSTRIDE; i += 128) lb[i] = (bf16_t)0.0f;
  for (int i = tid; i < MT * CL; i += 128) win[i / CL][i % CL] = 0.0f;
  for (int i = tid; i < CL * CL; i += 128) { wc1s[i] = Wc1[i]; wc2s[i] = Wc2[i]; }
  if (tid < CL) { bc1s[tid] = bc1[tid]; bc2s[tid] = bc2[tid]; }

  float pm = 0.0f;
  __syncthreads();

  for (int n = 0; n < NNOTES; ++n) {
    if (tid < MT) {
      float w[CL], l1[CL];
#pragma unroll
      for (int k = 0; k < CL; ++k) w[k] = win[tid][k];
#pragma unroll
      for (int j = 0; j < CL; ++j) {
        float s = bc1s[j];
#pragma unroll
        for (int k = 0; k < CL; ++k) s += wc1s[j * CL + k] * w[k];
        l1[j] = fmaxf(s, 0.0f);
      }
#pragma unroll
      for (int j = 0; j < CL; ++j) {
        float s = bc2s[j];
#pragma unroll
        for (int k = 0; k < CL; ++k) s += wc2s[j * CL + k] * l1[k];
        lb[tid * LSTRIDE + j] = (bf16_t)s;
      }
    }
    __syncthreads();

    v8f acc[16];
#pragma unroll
    for (int t = 0; t < 16; ++t) acc[t] = v8f_zero();

    const float* w1n = W1 + (size_t)n * EMB * DTOT;

    for (int kb = 0; kb < 16; ++kb) {
      const bf16_t* ap = &xl[(wave * 16 + m16) * XSTRIDE + kb * 32 + hi * 8];
      v8bf c0 = *(const v8bf*)ap;
      v8bf c1 = *(const v8bf*)(ap + 16);
      v16bf a;
#pragma unroll
      for (int i = 0; i < 8; ++i) { a[i] = c0[i]; a[i + 8] = c1[i]; }
#pragma unroll
      for (int nb = 0; nb < 16; ++nb) {
        const float* bp = w1n + (size_t)(nb * 16 + m16) * DTOT + kb * 32 + hi * 8;
        v16bf bfrag;
#pragma unroll
        for (int i = 0; i < 4; ++i) {
          float4 f = ((const float4*)(bp + (i >> 1) * 8))[i];
          bfrag[4 * i + 0] = (bf16_t)f.x;
          bfrag[4 * i + 1] = (bf16_t)f.y;
          bfrag[4 * i + 2] = (bf16_t)f.z;
          bfrag[4 * i + 3] = (bf16_t)f.w;
        }
        acc[nb] = __builtin_amdgcn_wmma_f32_16x16x32_bf16(
            false, a, false, bfrag, (short)0, acc[nb], false, false);
      }
    }
    {
      const bf16_t* ap = &lb[(wave * 16 + m16) * LSTRIDE + hi * 8];
      v8bf c0 = *(const v8bf*)ap;
      v8bf c1 = *(const v8bf*)(ap + 16);
      v16bf a;
#pragma unroll
      for (int i = 0; i < 8; ++i) { a[i] = c0[i]; a[i + 8] = c1[i]; }
#pragma unroll
      for (int nb = 0; nb < 16; ++nb) {
        const float* bp = w1n + (size_t)(nb * 16 + m16) * DTOT + XDIM;
        v16bf bfrag;
#pragma unroll
        for (int i = 0; i < 16; ++i) {
          int kk = (i < 8) ? hi * 8 + i : 8 + hi * 8 + i;
          float v = (kk < CL) ? bp[kk] : 0.0f;
          bfrag[i] = (bf16_t)v;
        }
        acc[nb] = __builtin_amdgcn_wmma_f32_16x16x32_bf16(
            false, a, false, bfrag, (short)0, acc[nb], false, false);
      }
    }

    float part[8];
#pragma unroll
    for (int r = 0; r < 8; ++r) part[r] = 0.0f;
#pragma unroll
    for (int nb = 0; nb < 16; ++nb) {
      int e = nb * 16 + m16;
      float b1v = b1[n * EMB + e];
      float w2v = W2[n * EMB + e];
#pragma unroll
      for (int r = 0; r < 8; ++r) {
        float av = fmaxf(acc[nb][r] + b1v, 0.0f);
        part[r] += av * w2v;
      }
    }
#pragma unroll
    for (int r = 0; r < 8; ++r) {
#pragma unroll
      for (int off = 1; off < 16; off <<= 1)
        part[r] += __shfl_xor(part[r], off, 32);
    }
    if (m16 == 0) {
      float b2v = b2[n];
#pragma unroll
      for (int r = 0; r < 8; ++r)
        logits[wave * 16 + hi * 8 + r] = part[r] + b2v;
    }
    __syncthreads();

    if (tid < MT) {
      int gp = rowbase + tid;
      float logit = logits[tid];
      float prob  = 1.0f / (1.0f + __expf(-logit));
      float u     = noise[(size_t)n * BSTOT + gp];
      float pred  = (u < prob) ? 1.0f : 0.0f;
      pm += prob * pred + (1.0f - prob) * (1.0f - pred);
#pragma unroll
      for (int k = 0; k < CL - 1; ++k) win[tid][k] = win[tid][k + 1];
      win[tid][CL - 1] = pred;
      labels[(size_t)gp * NNOTES + n] = pred;
    }
  }

  if (tid < MT) pmass[rowbase + tid] = pm;
}

extern "C" void kernel_launch(void* const* d_in, const int* in_sizes, int n_in,
                              void* d_out, int out_size, void* d_ws, size_t ws_size,
                              hipStream_t stream) {
  (void)in_sizes; (void)n_in; (void)out_size;
  const float* x    = (const float*)d_in[0];
  const float* Wc1  = (const float*)d_in[1];
  const float* bc1  = (const float*)d_in[2];
  const float* Wc2  = (const float*)d_in[3];
  const float* bc2  = (const float*)d_in[4];
  const float* W1   = (const float*)d_in[5];
  const float* b1   = (const float*)d_in[6];
  const float* W2   = (const float*)d_in[7];
  const float* b2   = (const float*)d_in[8];
  const float* nz   = (const float*)d_in[9];
  float* labels = (float*)d_out;                       // [8192, 88]
  float* pmass  = labels + (size_t)BSTOT * NNOTES;     // [8192]

  const size_t w1bf_bytes = W1BF_ELEMS * sizeof(bf16_t);   // ~24.5 MB
  dim3 grid(BSTOT / MT);   // 128 workgroups
  dim3 block(128);         // 4 wave32 per workgroup

  if (ws_size >= w1bf_bytes) {
    bf16_t* w1bf = (bf16_t*)d_ws;
    int cv_blocks = (int)((W1BF_ELEMS + 255) / 256);
    convert_w1_kernel<<<dim3(cv_blocks), dim3(256), 0, stream>>>(W1, w1bf);
    chain_wmma_pre<<<grid, block, 0, stream>>>(
        x, Wc1, bc1, Wc2, bc2, w1bf, b1, W2, b2, nz, labels, pmass);
  } else {
    chain_wmma_fb<<<grid, block, 0, stream>>>(
        x, Wc1, bc1, Wc2, bc2, W1, b1, W2, b2, nz, labels, pmass);
  }
}
